// GTFBlock_22711787061765
// MI455X (gfx1250) — matrix operand order repaired and checked by reading, not compiled, not run on hardware
//
#include <hip/hip_runtime.h>
#include <math.h>

// ---------------- problem constants ----------------
#define L_NODES 8192
#define KNBR    32
#define LKROWS  (L_NODES * KNBR)   // 262144
#define DN      128
#define DE      64
#define NH      8
#define HD      16                 // KD == VD == 16
#define VDE     8
#define DATTN   192
#define BN_EPS  1e-5f
#define SCALE_NODE 0.17677669529663687f   // (2*16)^-0.5
#define SCALE_EDGE 0.70710678118654752f   // 2^-0.5

typedef _Float16 f16_t;
typedef __attribute__((ext_vector_type(16))) _Float16 v16h;
typedef __attribute__((ext_vector_type(8)))  _Float16 v8h;
typedef __attribute__((ext_vector_type(8)))  float    v8f;

__device__ __forceinline__ v8f wmma16x16x32(v16h a, v16h b, v8f c) {
  // D = A(16x32 f16) * B(32x16 f16) + C(16x16 f32)
  return __builtin_amdgcn_wmma_f32_16x16x32_f16(false, a, false, b, (short)0, c,
                                                false, false);
}

// ---------------- generic WMMA GEMM -----------------
// Y[M x N] = f(A[M x F]) @ W[F x N] (+bias) (+Rres), N <= NT*16, F % 32 == 0.
// f = optional per-feature affine (BN fold, compile-time) then optional relu.
// Loop order: K outer (LDS-staged 32 x N f16 weight slab, shared by all 4
// waves of the block), n-tile inner with NT resident accumulator tiles, so
// every A K-slice is loaded exactly once as two float4s per lane.
// MODE 0: A = X[M x F].  MODE 1: A row r=(i*32+j) is the in-register concat
// [ xs[i](128) | ev[r](64) | xs[idx[r]](128) ]; segment boundaries (128,192)
// are 32-aligned so each K-step hits exactly one segment.
template <int MODE, int NT, bool BN, bool RELU, bool RESID, bool BIAS>
__global__ __launch_bounds__(128)
void gemm_wmma_kernel(const float* __restrict__ X, int M, int F,
                      const float* __restrict__ W, int N,
                      const float* __restrict__ bias,
                      const float* __restrict__ scl,
                      const float* __restrict__ shf,
                      const float* __restrict__ Rres,
                      float* __restrict__ Y,
                      const float* __restrict__ xs,
                      const float* __restrict__ ev,
                      const int* __restrict__ idx) {
  __shared__ __align__(16) f16_t wlds[NT * 16 * 32];  // col-major [n][k0..31]

  const int lane  = threadIdx.x & 31;
  const int wid   = threadIdx.x >> 5;
  const int mt    = blockIdx.x * (blockDim.x >> 5) + wid;
  const int m0    = mt * 16;
  const bool active = (m0 < M);        // wave-uniform; must still hit barriers
  const int r     = m0 + (lane & 15);  // A row for this lane
  const int khalf = lane >> 4;         // A k-half / D row-half select
  const int c     = lane & 15;         // B/D column for this lane

  int i_src = 0, i_trg = 0;
  if (MODE == 1) {
    i_src = r >> 5;
    i_trg = active ? idx[r] : 0;
  }

  v8f acc[NT];
#pragma unroll
  for (int nt = 0; nt < NT; ++nt) acc[nt] = (v8f){};

  const int NW = NT * 16;
  for (int kb = 0; kb < F; kb += 32) {
    // ---- cooperative stage of W[kb:kb+32, 0:N] into LDS (f16, col-major) ----
    __syncthreads();
    for (int t = threadIdx.x; t < 32 * NW; t += (int)blockDim.x) {
      const int n  = t % NW;           // consecutive threads -> consecutive n
      const int kr = t / NW;           //   => coalesced row-major W reads
      const float wv = (n < N) ? W[(size_t)(kb + kr) * N + n] : 0.0f;
      wlds[n * 32 + kr] = (f16_t)wv;
    }
    __syncthreads();
    if (!active) continue;

    // ---- A 16x32 f16 tile: lanes 0-15 rows, K 0-7 & 16-23; lanes 16-31: 8-15 & 24-31
    v16h a;
#pragma unroll
    for (int half = 0; half < 2; ++half) {
      const int ks = kb + half * 16 + khalf * 8;  // 8-aligned run, one segment
      const float* ap;
      if (MODE == 0) {
        ap = X + (size_t)r * F + ks;
      } else {
        if (ks < 128)      ap = xs + (size_t)i_src * 128 + ks;
        else if (ks < 192) ap = ev + (size_t)r * 64 + (ks - 128);
        else               ap = xs + (size_t)i_trg * 128 + (ks - 192);
      }
      const float4 x0 = *(const float4*)(ap);
      const float4 x1 = *(const float4*)(ap + 4);
      float xv[8] = {x0.x, x0.y, x0.z, x0.w, x1.x, x1.y, x1.z, x1.w};
      if (BN) {
        const float4 s0 = *(const float4*)(scl + ks);
        const float4 s1 = *(const float4*)(scl + ks + 4);
        const float4 h0 = *(const float4*)(shf + ks);
        const float4 h1 = *(const float4*)(shf + ks + 4);
        const float sv[8] = {s0.x, s0.y, s0.z, s0.w, s1.x, s1.y, s1.z, s1.w};
        const float hv[8] = {h0.x, h0.y, h0.z, h0.w, h1.x, h1.y, h1.z, h1.w};
#pragma unroll
        for (int e = 0; e < 8; ++e) xv[e] = xv[e] * sv[e] + hv[e];
      }
#pragma unroll
      for (int e = 0; e < 8; ++e) {
        float v = xv[e];
        if (RELU) v = fmaxf(v, 0.0f);
        a[half * 8 + e] = (f16_t)v;
      }
    }

    // ---- B from LDS + NT WMMAs ----
#pragma unroll
    for (int nt = 0; nt < NT; ++nt) {
      const f16_t* bp = &wlds[(nt * 16 + c) * 32 + khalf * 8];
      const v8h b0 = *(const v8h*)(bp);
      const v8h b1 = *(const v8h*)(bp + 16);
      v16h b;
#pragma unroll
      for (int e = 0; e < 8; ++e) { b[e] = b0[e]; b[8 + e] = b1[e]; }
      acc[nt] = wmma16x16x32(a, b, acc[nt]);
    }
  }

  // ---- store: lane holds col n, rows m0 + khalf*8 + v ----
  if (active) {
#pragma unroll
    for (int nt = 0; nt < NT; ++nt) {
      const int n = nt * 16 + c;
      if (n < N) {
        const float bv = BIAS ? bias[n] : 0.0f;
#pragma unroll
        for (int v = 0; v < 8; ++v) {
          const int m = m0 + khalf * 8 + v;
          float yv = acc[nt][v] + bv;
          if (RESID) yv += Rres[(size_t)m * N + n];
          Y[(size_t)m * N + n] = yv;
        }
      }
    }
  }
}

// ---------------- BN statistics ---------------------
__global__ void zero_kernel(float* __restrict__ p, int n) {
  int i = blockIdx.x * blockDim.x + threadIdx.x;
  for (; i < n; i += gridDim.x * blockDim.x) p[i] = 0.0f;
}

// blockDim.x == F; each block reduces a chunk of rows, then atomics.
__global__ void bn_stats_kernel(const float* __restrict__ X, int nrows, int F,
                                float* __restrict__ sum, float* __restrict__ sumsq,
                                int rows_per_block) {
  const int f = threadIdx.x;
  int r  = blockIdx.x * rows_per_block;
  int re = r + rows_per_block;
  if (re > nrows) re = nrows;
  float s = 0.0f, s2 = 0.0f;
  for (; r < re; ++r) {
    float x = X[(size_t)r * F + f];
    s += x;
    s2 += x * x;
  }
  atomicAdd(&sum[f], s);
  atomicAdd(&sumsq[f], s2);
}

__global__ void bn_finalize_kernel(const float* __restrict__ sum,
                                   const float* __restrict__ sumsq,
                                   const float* __restrict__ g,
                                   const float* __restrict__ b,
                                   float inv_n, int F,
                                   float* __restrict__ scl, float* __restrict__ shf) {
  int f = blockIdx.x * blockDim.x + threadIdx.x;
  if (f < F) {
    float m = sum[f] * inv_n;
    float v = sumsq[f] * inv_n - m * m;
    float sc = g[f] * rsqrtf(v + BN_EPS);
    scl[f] = sc;
    shf[f] = b[f] - m * sc;
  }
}

// y = x*scale[f] + shift[f]   (elementwise BN apply; materializes xn)
__global__ void affine_kernel(const float* __restrict__ X, long n, int F,
                              const float* __restrict__ scl,
                              const float* __restrict__ shf,
                              float* __restrict__ Y) {
  long i = (long)blockIdx.x * blockDim.x + threadIdx.x;
  for (; i < n; i += (long)gridDim.x * blockDim.x) {
    int f = (int)(i % F);
    Y[i] = X[i] * scl[f] + shf[f];
  }
}

// ev_out = ev + relu(bn(e))
__global__ void ev_update_kernel(const float* __restrict__ ev,
                                 const float* __restrict__ e, long n, int F,
                                 const float* __restrict__ scl,
                                 const float* __restrict__ shf,
                                 float* __restrict__ out) {
  long i = (long)blockIdx.x * blockDim.x + threadIdx.x;
  for (; i < n; i += (long)gridDim.x * blockDim.x) {
    int f = (int)(i % F);
    out[i] = ev[i] + fmaxf(e[i] * scl[f] + shf[f], 0.0f);
  }
}

// in-place rotary on Q and K: thread per (l, h, pair)
__global__ void rotary_kernel(float* __restrict__ Q, float* __restrict__ Km) {
  int t = blockIdx.x * blockDim.x + threadIdx.x;
  if (t >= L_NODES * NH * (HD / 2)) return;
  int p = t & 7;
  int h = (t >> 3) & 7;
  int l = t >> 6;
  float inv = powf(10000.0f, -(float)(2 * p) / (float)HD);
  float ang = (float)l * inv;
  float cs = cosf(ang), sn = sinf(ang);
  int base = l * (NH * HD) + h * HD + 2 * p;
  float a = Q[base], b = Q[base + 1];
  Q[base]     = a * cs - b * sn;
  Q[base + 1] = b * cs + a * sn;
  a = Km[base]; b = Km[base + 1];
  Km[base]     = a * cs - b * sn;
  Km[base + 1] = b * cs + a * sn;
}

// gather attention: thread per (node i, head h)
__global__ void attn_kernel(const float* __restrict__ Q, const float* __restrict__ Km,
                            const float* __restrict__ V,
                            const float* __restrict__ eb,      // [LK x 8]
                            const float* __restrict__ v_edge,  // [LK x 64]
                            const int* __restrict__ idx,
                            float* __restrict__ results) {     // [L x 192]
  int t = blockIdx.x * blockDim.x + threadIdx.x;
  if (t >= L_NODES * NH) return;
  int h = t & 7;
  int i = t >> 3;
  float q[HD];
#pragma unroll
  for (int d = 0; d < HD; ++d) q[d] = Q[i * DN + h * HD + d];

  int nb[KNBR];
  float lg[KNBR];
  float mx = -3.0e38f;
#pragma unroll
  for (int j = 0; j < KNBR; ++j) {
    nb[j] = idx[i * KNBR + j];
    const float* kr = Km + (size_t)nb[j] * DN + h * HD;
    float dot = 0.0f;
#pragma unroll
    for (int d = 0; d < HD; d += 4) {
      const float4 kv = *(const float4*)(kr + d);
      dot += q[d] * kv.x + q[d + 1] * kv.y + q[d + 2] * kv.z + q[d + 3] * kv.w;
    }
    lg[j] = dot * SCALE_NODE + eb[(size_t)(i * KNBR + j) * NH + h] * SCALE_EDGE;
    mx = fmaxf(mx, lg[j]);
  }
  float ssum = 0.0f;
#pragma unroll
  for (int j = 0; j < KNBR; ++j) {
    lg[j] = expf(lg[j] - mx);
    ssum += lg[j];
  }
  float inv = 1.0f / ssum;
  float rn[HD];
  float re[VDE];
#pragma unroll
  for (int d = 0; d < HD; ++d) rn[d] = 0.0f;
#pragma unroll
  for (int d = 0; d < VDE; ++d) re[d] = 0.0f;
#pragma unroll
  for (int j = 0; j < KNBR; ++j) {
    const float w = lg[j] * inv;
    const float* vr = V + (size_t)nb[j] * DN + h * HD;
#pragma unroll
    for (int d = 0; d < HD; d += 4) {
      const float4 vv = *(const float4*)(vr + d);
      rn[d] += w * vv.x; rn[d + 1] += w * vv.y;
      rn[d + 2] += w * vv.z; rn[d + 3] += w * vv.w;
    }
    const float* ver = v_edge + (size_t)(i * KNBR + j) * DE + h * VDE;
#pragma unroll
    for (int d = 0; d < VDE; d += 4) {
      const float4 vv = *(const float4*)(ver + d);
      re[d] += w * vv.x; re[d + 1] += w * vv.y;
      re[d + 2] += w * vv.z; re[d + 3] += w * vv.w;
    }
  }
#pragma unroll
  for (int d = 0; d < HD; ++d) results[(size_t)i * DATTN + h * HD + d] = rn[d];
#pragma unroll
  for (int d = 0; d < VDE; ++d)
    results[(size_t)i * DATTN + NH * HD + h * VDE + d] = re[d];
}

// ---------------- host orchestration ----------------
static inline dim3 gemm_grid(int M) { return dim3((unsigned)((M / 16 + 3) / 4)); }

extern "C" void kernel_launch(void* const* d_in, const int* in_sizes, int n_in,
                              void* d_out, int out_size, void* d_ws, size_t ws_size,
                              hipStream_t stream) {
  const float* x        = (const float*)d_in[0];
  const float* edgevec  = (const float*)d_in[1];
  const int*   nbr      = (const int*)d_in[2];
  const float* eu0_w    = (const float*)d_in[3];
  const float* eu0_b    = (const float*)d_in[4];
  // resblock params: eu1_ at 5..12, eu2_ at 13..20
  const float* eu_bn_g  = (const float*)d_in[21];
  const float* eu_bn_b  = (const float*)d_in[22];
  const float* an_g     = (const float*)d_in[23];
  const float* an_b     = (const float*)d_in[24];
  const float* wq       = (const float*)d_in[25];
  const float* wk       = (const float*)d_in[26];
  const float* wv       = (const float*)d_in[27];
  const float* web_w    = (const float*)d_in[28];
  const float* web_b    = (const float*)d_in[29];
  const float* wev_w    = (const float*)d_in[30];
  const float* wev_b    = (const float*)d_in[31];
  const float* ffn_g    = (const float*)d_in[32];
  const float* ffn_b    = (const float*)d_in[33];
  const float* ff1_w    = (const float*)d_in[34];
  const float* ff1_b    = (const float*)d_in[35];
  const float* ff2_w    = (const float*)d_in[36];
  const float* ff2_b    = (const float*)d_in[37];
  const float* to_out_w = (const float*)d_in[38];
  const float* to_out_b = (const float*)d_in[39];

  float* out_node = (float*)d_out;                        // [L x 128]
  float* ev_out   = (float*)d_out + (size_t)L_NODES * DN; // [L x K x 64]

  // workspace layout (floats)
  float* ws  = (float*)d_ws;
  float* E   = ws;                          // LK*64
  float* Hb  = ws + (size_t)16777216;       // LK*64 (resblock tmp, later v_edge)
  float* XN  = ws + (size_t)33554432;       // L*128
  float* Qb  = ws + (size_t)34603008;       // L*128
  float* Kb  = ws + (size_t)35651584;       // L*128
  float* Vb  = ws + (size_t)36700160;       // L*128
  float* EB  = ws + (size_t)37748736;       // LK*8
  float* RES = ws + (size_t)39845888;       // L*192
  float* ST  = ws + (size_t)41418752;       // stats scratch
  float* SUM = ST, *SSQ = ST + 192, *SCL = ST + 384, *SHF = ST + 576;
  float* T1 = E;                            // L*192 (edge buffers free by then)
  float* T2 = E + (size_t)1572864;

  const int RPB = 512;  // rows per stats block
  auto bn_pass = [&](const float* Xb, int nrows, int F, const float* g,
                     const float* b) {
    zero_kernel<<<2, 256, 0, stream>>>(ST, 384);
    bn_stats_kernel<<<dim3((nrows + RPB - 1) / RPB), F, 0, stream>>>(
        Xb, nrows, F, SUM, SSQ, RPB);
    bn_finalize_kernel<<<1, 256, 0, stream>>>(SUM, SSQ, g, b,
                                              1.0f / (float)nrows, F, SCL, SHF);
  };

  // ---- node BN (an) + xn + QKV + rotary ----
  bn_pass(x, L_NODES, DN, an_g, an_b);
  affine_kernel<<<1024, 256, 0, stream>>>(x, (long)L_NODES * DN, DN, SCL, SHF, XN);
  gemm_wmma_kernel<0, 8, false, false, false, false>
      <<<gemm_grid(L_NODES), 128, 0, stream>>>(XN, L_NODES, DN, wq, DN, nullptr,
                                               nullptr, nullptr, nullptr, Qb,
                                               nullptr, nullptr, nullptr);
  gemm_wmma_kernel<0, 8, false, false, false, false>
      <<<gemm_grid(L_NODES), 128, 0, stream>>>(XN, L_NODES, DN, wk, DN, nullptr,
                                               nullptr, nullptr, nullptr, Kb,
                                               nullptr, nullptr, nullptr);
  gemm_wmma_kernel<0, 8, false, false, false, false>
      <<<gemm_grid(L_NODES), 128, 0, stream>>>(XN, L_NODES, DN, wv, DN, nullptr,
                                               nullptr, nullptr, nullptr, Vb,
                                               nullptr, nullptr, nullptr);
  rotary_kernel<<<(L_NODES * NH * (HD / 2) + 255) / 256, 256, 0, stream>>>(Qb, Kb);

  // ---- edge MLP: e0 = [x_src | ev | x_trg] @ eu0_w + b (in-register concat) ----
  gemm_wmma_kernel<1, 4, false, false, false, true>
      <<<gemm_grid(LKROWS), 128, 0, stream>>>(nullptr, LKROWS, 2 * DN + DE,
                                              eu0_w, DE, eu0_b, nullptr, nullptr,
                                              nullptr, E, x, edgevec, nbr);

  // ---- two residual blocks on E ----
  for (int rb = 0; rb < 2; ++rb) {
    int base = 5 + rb * 8;
    const float* bn1_g = (const float*)d_in[base + 0];
    const float* bn1_b = (const float*)d_in[base + 1];
    const float* l1_w  = (const float*)d_in[base + 2];
    const float* l1_b  = (const float*)d_in[base + 3];
    const float* bn2_g = (const float*)d_in[base + 4];
    const float* bn2_b = (const float*)d_in[base + 5];
    const float* l2_w  = (const float*)d_in[base + 6];
    const float* l2_b  = (const float*)d_in[base + 7];
    bn_pass(E, LKROWS, DE, bn1_g, bn1_b);
    gemm_wmma_kernel<0, 4, true, true, false, true>
        <<<gemm_grid(LKROWS), 128, 0, stream>>>(E, LKROWS, DE, l1_w, DE, l1_b,
                                                SCL, SHF, nullptr, Hb,
                                                nullptr, nullptr, nullptr);
    bn_pass(Hb, LKROWS, DE, bn2_g, bn2_b);
    gemm_wmma_kernel<0, 4, true, true, true, true>
        <<<gemm_grid(LKROWS), 128, 0, stream>>>(Hb, LKROWS, DE, l2_w, DE, l2_b,
                                                SCL, SHF, /*residual=*/E, E,
                                                nullptr, nullptr, nullptr);
  }

  // ---- ev_out = ev + relu(bn(E)) ----
  bn_pass(E, LKROWS, DE, eu_bn_g, eu_bn_b);
  ev_update_kernel<<<2048, 256, 0, stream>>>(edgevec, E, (long)LKROWS * DE, DE,
                                             SCL, SHF, ev_out);

  // ---- edge bias + edge values from updated ev ----
  gemm_wmma_kernel<0, 1, false, false, false, true>
      <<<gemm_grid(LKROWS), 128, 0, stream>>>(ev_out, LKROWS, DE, web_w, NH,
                                              web_b, nullptr, nullptr, nullptr,
                                              EB, nullptr, nullptr, nullptr);
  gemm_wmma_kernel<0, 4, false, false, false, true>
      <<<gemm_grid(LKROWS), 128, 0, stream>>>(ev_out, LKROWS, DE, wev_w, DE,
                                              wev_b, nullptr, nullptr, nullptr,
                                              Hb, nullptr, nullptr, nullptr);

  // ---- gather attention ----
  attn_kernel<<<(L_NODES * NH + 255) / 256, 256, 0, stream>>>(Qb, Kb, Vb, EB, Hb,
                                                              nbr, RES);

  // ---- output FFN: bn -> ff1 -> relu -> ff2 -> to_out (+xn residual) ----
  bn_pass(RES, L_NODES, DATTN, ffn_g, ffn_b);
  gemm_wmma_kernel<0, 12, true, false, false, true>
      <<<gemm_grid(L_NODES), 128, 0, stream>>>(RES, L_NODES, DATTN, ff1_w, DATTN,
                                               ff1_b, SCL, SHF, nullptr, T1,
                                               nullptr, nullptr, nullptr);
  gemm_wmma_kernel<0, 12, false, true, false, true>
      <<<gemm_grid(L_NODES), 128, 0, stream>>>(T1, L_NODES, DATTN, ff2_w, DATTN,
                                               ff2_b, nullptr, nullptr, nullptr,
                                               T2, nullptr, nullptr, nullptr);
  gemm_wmma_kernel<0, 8, false, false, true, true>
      <<<gemm_grid(L_NODES), 128, 0, stream>>>(T2, L_NODES, DATTN, to_out_w, DN,
                                               to_out_b, nullptr, nullptr,
                                               /*residual=*/XN, out_node,
                                               nullptr, nullptr, nullptr);

  (void)in_sizes; (void)n_in; (void)out_size; (void)ws_size;
}
